// GRU_962072674355
// MI455X (gfx1250) — compile-verified, matrix operand order checked by reading, not run on hardware
//
#include <hip/hip_runtime.h>
#include <hip/hip_bf16.h>

// ---------------------------------------------------------------------------
// GRU encoder/decoder for MI455X (gfx1250, wave32).
// Strategy: split-bf16 (hi+lo) WMMA GEMMs with f32 accumulate:
//   A*B ~= Ahi*Bhi + Ahi*Blo + Alo*Bhi   (3x v_wmma_f32_16x16x32_bf16)
// ~18 effective mantissa bits, adequate for a 151-step recurrence.
// Weights (bf16 hi+lo 16MiB) stay resident in the 192MB L2 across steps.
// WMMAs are issued product-type-major across 4 independent accumulators so
// consecutive WMMAs never chain on the same D/C registers (avoids the XDL
// WMMA->WMMA hazard nops).
// ---------------------------------------------------------------------------

typedef __attribute__((ext_vector_type(16))) __bf16 bf16x16;
typedef __attribute__((ext_vector_type(8)))  __bf16 bf16x8;
typedef __attribute__((ext_vector_type(8)))  float  f32x8;

union Frag16 { bf16x16 v; bf16x8 h[2]; };

__device__ __forceinline__ unsigned short f32_to_bf16_bits(float f) {
  unsigned int u = __float_as_uint(f);
  u += 0x7FFFu + ((u >> 16) & 1u);            // round-to-nearest-even
  return (unsigned short)(u >> 16);
}
__device__ __forceinline__ float bf16_bits_to_f32(unsigned short b) {
  return __uint_as_float(((unsigned int)b) << 16);
}
__device__ __forceinline__ void split2bf16(float f, __bf16* hi, __bf16* lo) {
  unsigned short hb = f32_to_bf16_bits(f);
  float hf = bf16_bits_to_f32(hb);
  unsigned short lb = f32_to_bf16_bits(f - hf);
  union { unsigned short u; __bf16 b; } ch, cl;
  ch.u = hb; cl.u = lb;
  *hi = ch.b; *lo = cl.b;
}

// ---------------------------------------------------------------------------
// C[M x N] = A[M x K] * W[N x K]^T + bias, split-bf16 inputs, f32 out.
// Workgroup: 256 threads = 8 wave32s, tile 32(M) x 256(N).
// Wave (wm,wn): wm in {0,1} x 16 rows, wn in {0..3} x 64 cols (4 WMMA tiles).
// A-fragment (16-bit, 16x32 MxK): lane l -> row l%16, K chunks at
//   (l/16)*8 and 16+(l/16)*8 (two 16B loads). W row-major == B^T, so B
// fragments load with the identical pattern from row (col of C).
// C/D layout: acc[g] of lane l -> M = g + 8*(l>>4), N = l&15.
// ---------------------------------------------------------------------------
__global__ __launch_bounds__(256) void gemm_bf16x2_wmma(
    const __bf16* __restrict__ Ahi, const __bf16* __restrict__ Alo, int lda,
    const __bf16* __restrict__ Whi, const __bf16* __restrict__ Wlo,
    const float* __restrict__ bias,
    float* __restrict__ C, int ldc,
    __bf16* __restrict__ Ohi, __bf16* __restrict__ Olo, int ldo,
    int K)
{
  const int lane = threadIdx.x & 31;
  const int wave = threadIdx.x >> 5;
  const int wm   = wave & 1;
  const int wn   = wave >> 1;
  const int l16  = lane & 15;
  const int half = lane >> 4;
  const int rowBase = blockIdx.x * 32 + wm * 16;
  const int colBase = blockIdx.y * 256 + wn * 64;

  const __bf16* aHiP = Ahi + (size_t)(rowBase + l16) * lda;
  const __bf16* aLoP = Alo + (size_t)(rowBase + l16) * lda;

  const __bf16* wHiP[4];
  const __bf16* wLoP[4];
#pragma unroll
  for (int s = 0; s < 4; ++s) {
    wHiP[s] = Whi + (size_t)(colBase + s * 16 + l16) * K;
    wLoP[s] = Wlo + (size_t)(colBase + s * 16 + l16) * K;
  }

  f32x8 acc[4] = {};

  for (int k0 = 0; k0 < K; k0 += 32) {
    Frag16 ah, al;
    ah.h[0] = *(const bf16x8*)(aHiP + k0 + half * 8);
    ah.h[1] = *(const bf16x8*)(aHiP + k0 + 16 + half * 8);
    al.h[0] = *(const bf16x8*)(aLoP + k0 + half * 8);
    al.h[1] = *(const bf16x8*)(aLoP + k0 + 16 + half * 8);
    __builtin_prefetch(aHiP + k0 + 64, 0, 3);   // global_prefetch_b8 (near)

    Frag16 bh[4], bl[4];
#pragma unroll
    for (int s = 0; s < 4; ++s) {
      bh[s].h[0] = *(const bf16x8*)(wHiP[s] + k0 + half * 8);
      bh[s].h[1] = *(const bf16x8*)(wHiP[s] + k0 + 16 + half * 8);
      bl[s].h[0] = *(const bf16x8*)(wLoP[s] + k0 + half * 8);
      bl[s].h[1] = *(const bf16x8*)(wLoP[s] + k0 + 16 + half * 8);
    }

    // Product-type-major issue order: consecutive WMMAs hit different
    // accumulators -> no back-to-back D/C chaining hazards.
#pragma unroll
    for (int s = 0; s < 4; ++s)
      acc[s] = __builtin_amdgcn_wmma_f32_16x16x32_bf16(
          false, ah.v, false, bl[s].v, (short)0, acc[s], false, false);
#pragma unroll
    for (int s = 0; s < 4; ++s)
      acc[s] = __builtin_amdgcn_wmma_f32_16x16x32_bf16(
          false, al.v, false, bh[s].v, (short)0, acc[s], false, false);
#pragma unroll
    for (int s = 0; s < 4; ++s)
      acc[s] = __builtin_amdgcn_wmma_f32_16x16x32_bf16(
          false, ah.v, false, bh[s].v, (short)0, acc[s], false, false);
  }

#pragma unroll
  for (int s = 0; s < 4; ++s) {
    const int c = colBase + s * 16 + l16;
    const float bv = bias ? bias[c] : 0.0f;
#pragma unroll
    for (int g = 0; g < 8; ++g) {
      const int r = rowBase + g + 8 * half;
      const float v = acc[s][g] + bv;
      C[(size_t)r * ldc + c] = v;
      if (Ohi) {
        split2bf16(v, &Ohi[(size_t)r * ldo + c], &Olo[(size_t)r * ldo + c]);
      }
    }
  }
}

// ---------------------------------------------------------------------------
// Fused GRU gate nonlinearity + state update + bf16 hi/lo re-split of h.
// gi/gh already contain b_ih / b_hh from the GEMM epilogue.
// ---------------------------------------------------------------------------
__global__ __launch_bounds__(256) void gru_update_kernel(
    const float* __restrict__ gi, const float* __restrict__ gh,
    float* __restrict__ h, __bf16* __restrict__ hhi, __bf16* __restrict__ hlo)
{
  const int idx = blockIdx.x * 256 + threadIdx.x;   // < B*H
  const int b = idx >> 10;                          // H = 1024
  const int j = idx & 1023;
  const size_t gb = (size_t)b * 3072;
  const float ir = gi[gb + j], iz = gi[gb + 1024 + j], in_ = gi[gb + 2048 + j];
  const float hr = gh[gb + j], hz = gh[gb + 1024 + j], hn  = gh[gb + 2048 + j];
  const float r = 1.0f / (1.0f + __expf(-(ir + hr)));
  const float z = 1.0f / (1.0f + __expf(-(iz + hz)));
  const float n = tanhf(in_ + r * hn);
  const float hnew = (1.0f - z) * n + z * h[idx];
  h[idx] = hnew;
  split2bf16(hnew, &hhi[idx], &hlo[idx]);
}

__global__ __launch_bounds__(256) void split_f32_kernel(
    const float* __restrict__ src, __bf16* __restrict__ hi,
    __bf16* __restrict__ lo, long n)
{
  long i = (long)blockIdx.x * blockDim.x + threadIdx.x;
  const long stride = (long)gridDim.x * blockDim.x;
  for (; i < n; i += stride) split2bf16(src[i], &hi[i], &lo[i]);
}

__global__ __launch_bounds__(256) void zero_state_kernel(
    float* __restrict__ h, __bf16* __restrict__ hhi, __bf16* __restrict__ hlo,
    int n)
{
  const int i = blockIdx.x * 256 + threadIdx.x;
  if (i < n) {
    h[i] = 0.0f;
    union { unsigned short u; __bf16 b; } zz; zz.u = 0;
    hhi[i] = zz.b; hlo[i] = zz.b;
  }
}

// ---------------------------------------------------------------------------
extern "C" void kernel_launch(void* const* d_in, const int* in_sizes, int n_in,
                              void* d_out, int out_size, void* d_ws, size_t ws_size,
                              hipStream_t stream) {
  (void)in_sizes; (void)n_in; (void)out_size; (void)ws_size;

  constexpr int B = 512, S = 128, I = 256, H = 1024, PRE = 24;
  constexpr int G = 3 * H;          // 3072 gate columns

  const float* x    = (const float*)d_in[0];
  const float* W_ih = (const float*)d_in[1];
  const float* W_hh = (const float*)d_in[2];
  const float* b_ih = (const float*)d_in[3];
  const float* b_hh = (const float*)d_in[4];
  const float* W_fc = (const float*)d_in[5];
  const float* b_fc = (const float*)d_in[6];
  float* out = (float*)d_out;       // [B, PRE, I]

  // ---- workspace carve-out (256B aligned) --------------------------------
  char* p = (char*)d_ws;
  auto alloc = [&](size_t bytes) -> char* {
    char* r = p;
    p += (bytes + 255) & ~(size_t)255;
    return r;
  };
  __bf16* xhi    = (__bf16*)alloc((size_t)B * S * I * 2);
  __bf16* xlo    = (__bf16*)alloc((size_t)B * S * I * 2);
  __bf16* Wih_hi = (__bf16*)alloc((size_t)G * I * 2);
  __bf16* Wih_lo = (__bf16*)alloc((size_t)G * I * 2);
  __bf16* Whh_hi = (__bf16*)alloc((size_t)G * H * 2);
  __bf16* Whh_lo = (__bf16*)alloc((size_t)G * H * 2);
  __bf16* Wfc_hi = (__bf16*)alloc((size_t)I * H * 2);
  __bf16* Wfc_lo = (__bf16*)alloc((size_t)I * H * 2);
  float*  hbuf   = (float*)alloc((size_t)B * H * 4);
  __bf16* hhi    = (__bf16*)alloc((size_t)B * H * 2);
  __bf16* hlo    = (__bf16*)alloc((size_t)B * H * 2);
  float*  gi     = (float*)alloc((size_t)B * G * 4);
  float*  gh     = (float*)alloc((size_t)B * G * 4);
  __bf16* dec_hi = (__bf16*)alloc((size_t)B * I * 2);
  __bf16* dec_lo = (__bf16*)alloc((size_t)B * I * 2);

  const dim3 blk(256);

  // ---- one-time bf16 hi/lo splits ----------------------------------------
  split_f32_kernel<<<2048, blk, 0, stream>>>(x,    xhi,    xlo,    (long)B * S * I);
  split_f32_kernel<<<1024, blk, 0, stream>>>(W_ih, Wih_hi, Wih_lo, (long)G * I);
  split_f32_kernel<<<2048, blk, 0, stream>>>(W_hh, Whh_hi, Whh_lo, (long)G * H);
  split_f32_kernel<<<512,  blk, 0, stream>>>(W_fc, Wfc_hi, Wfc_lo, (long)I * H);
  zero_state_kernel<<<(B * H) / 256, blk, 0, stream>>>(hbuf, hhi, hlo, B * H);

  auto gemm = [&](const __bf16* Ahi, const __bf16* Alo, int lda,
                  const __bf16* Whi, const __bf16* Wlo, const float* bias,
                  float* C, int ldc, __bf16* Ohi, __bf16* Olo, int ldo,
                  int M, int N, int K) {
    dim3 grid(M / 32, N / 256);
    gemm_bf16x2_wmma<<<grid, blk, 0, stream>>>(Ahi, Alo, lda, Whi, Wlo, bias,
                                               C, ldc, Ohi, Olo, ldo, K);
  };

  const int upd_blocks = (B * H) / 256;

  // ---- encoder: 128 sequential GRU steps ---------------------------------
  for (int t = 0; t < S; ++t) {
    gemm(xhi + (size_t)t * I, xlo + (size_t)t * I, S * I,
         Wih_hi, Wih_lo, b_ih, gi, G, nullptr, nullptr, 0, B, G, I);
    gemm(hhi, hlo, H, Whh_hi, Whh_lo, b_hh, gh, G, nullptr, nullptr, 0, B, G, H);
    gru_update_kernel<<<upd_blocks, blk, 0, stream>>>(gi, gh, hbuf, hhi, hlo);
  }

  // ---- out0 = h @ W_fc^T + b_fc  -> d_out[:,0,:] and bf16 feedback -------
  gemm(hhi, hlo, H, Wfc_hi, Wfc_lo, b_fc, out, PRE * I,
       dec_hi, dec_lo, I, B, I, H);

  // ---- autoregressive decoder --------------------------------------------
  for (int t = 1; t < PRE; ++t) {
    gemm(dec_hi, dec_lo, I, Wih_hi, Wih_lo, b_ih, gi, G,
         nullptr, nullptr, 0, B, G, I);
    gemm(hhi, hlo, H, Whh_hi, Whh_lo, b_hh, gh, G,
         nullptr, nullptr, 0, B, G, H);
    gru_update_kernel<<<upd_blocks, blk, 0, stream>>>(gi, gh, hbuf, hhi, hlo);
    gemm(hhi, hlo, H, Wfc_hi, Wfc_lo, b_fc, out + (size_t)t * I, PRE * I,
         dec_hi, dec_lo, I, B, I, H);
  }
}
